// DEVS_SO3_25228637897337
// MI455X (gfx1250) — matrix operand order repaired and checked by reading, not compiled, not run on hardware
//
#include <hip/hip_runtime.h>
#include <hip/hip_bf16.h>
#include <math.h>

// ---------------------------------------------------------------------------
// Problem constants (shapes fixed by the reference)
// ---------------------------------------------------------------------------
#define BSZ   64
#define SLEN  16
#define DIM   37632          // 3*112*112
#define HID   512
#define MROWS (BSZ*SLEN)     // 1024 encoder rows
#define DROWS (2*MROWS)      // 2048 decoder rows (z_enc ++ z_pred)
#define DT_C  0.001f
#define KSPLIT 4             // encoder FC1 split-K factor
#define KCHUNK (DIM/KSPLIT)  // 9408, divisible by 32

// Output regions (floats), reference return order
#define OUT_RECON  0
#define OUT_PRED   (MROWS*(size_t)DIM)
#define OUT_ZENC   (2*MROWS*(size_t)DIM)
#define OUT_ZPRED  (OUT_ZENC + MROWS*9)
#define OUT_PIENC  (OUT_ZPRED + MROWS*9)
#define OUT_PIPRED (OUT_PIENC + BSZ*15*3)

// Workspace layout (floats). Partials overlay hd (disjoint lifetimes).
#define WS_H    0                                   // h  : 1024 x 512
#define WS_Z    (MROWS*(size_t)HID)                 // z  : 2048 x 9
#define WS_HD   (WS_Z + DROWS*9)                    // hd : 2048 x 512
#define WS_PART WS_HD                               // P  : 4 x 1024 x 512

typedef __attribute__((ext_vector_type(16))) __bf16 v16bf;
typedef __attribute__((ext_vector_type(8)))  float  v8f;

#define TILE_M 128   // 8 waves x 16 rows
#define TILE_N 64    // 4 WMMA fragments wide
#define TILE_K 32

__device__ __forceinline__ unsigned short f2bf(float f) {
    unsigned int u = __float_as_uint(f);
    unsigned int r = (u + 0x7FFFu + ((u >> 16) & 1u)) >> 16;   // RNE
    return (unsigned short)r;
}

// Async DMA: 16 bytes global -> LDS per lane (ASYNCcnt-tracked, no VGPR data)
__device__ __forceinline__ void async_copy16(unsigned lds_off, const void* gptr) {
    unsigned long long g = (unsigned long long)(uintptr_t)gptr;
    asm volatile("global_load_async_to_lds_b128 %0, %1, off"
                 :: "v"(lds_off), "v"(g) : "memory");
}
__device__ __forceinline__ void wait_async0() {
    asm volatile("s_wait_asynccnt 0x0" ::: "memory");
}

// Stage one 32x64 fp32 B tile into LDS via async DMA (512 x 16B chunks).
__device__ __forceinline__ void stage_tile_async(
    const float* __restrict__ W, int N, int n0, int kt, unsigned ldsBase, int tid)
{
    #pragma unroll
    for (int h = 0; h < 2; ++h) {
        int c   = tid + h * 256;      // 0..511
        int row = c >> 4;             // 0..31
        int col = (c & 15) << 2;      // 0,4,..,60
        async_copy16(ldsBase + (unsigned)c * 16u,
                     W + (size_t)(kt + row) * N + n0 + col);
    }
}

// Convert one raw fp32 tile (LDS) into bf16 WMMA-fragment layout (LDS).
__device__ __forceinline__ void convert_tile(
    const float (*raw)[TILE_N], unsigned short (*frag)[32][16],
    int kS, int nbS, int hiK)
{
    const float4* rp = (const float4*)&raw[kS][nbS];
    float4 r0 = rp[0], r1 = rp[1];
    float wv[8] = {r0.x, r0.y, r0.z, r0.w, r1.x, r1.y, r1.z, r1.w};
    #pragma unroll
    for (int i = 0; i < 8; ++i) {
        int n = nbS + i;
        frag[n >> 4][(n & 15) | hiK][kS & 15] = f2bf(wv[i]);
    }
}

// ---------------------------------------------------------------------------
// bf16-WMMA GEMM over one K-chunk, async-LDS pipelined B staging.
//   act >= 0 : out = act(A@W + bias)  (0 = ELU, 1 = sigmoid), row split at
//              `split` redirects high rows to out1.
//   act <  0 : raw fp32 partials -> out0 + blockIdx.z * M*N  (split-K mode)
// ---------------------------------------------------------------------------
__global__ __launch_bounds__(256) void wmma_gemm(
    const float* __restrict__ A, const float* __restrict__ W,
    const float* __restrict__ bias,
    float* __restrict__ out0, float* __restrict__ out1,
    int M, int N, int K, int kchunk, int split, int act)
{
    __shared__ alignas(16) float          Raw[2][TILE_K][TILE_N];   // 2 x 8 KB
    __shared__ alignas(32) unsigned short Fsh[2][4][32][16];        // 2 x 4 KB

    const int tid  = threadIdx.x;
    const int wave = tid >> 5;
    const int lane = tid & 31;
    const int m0   = blockIdx.x * TILE_M + wave * 16;
    const int n0   = blockIdx.y * TILE_N;
    const int k0   = blockIdx.z * kchunk;
    const int k1   = k0 + kchunk;

    const unsigned rawBase0 = (unsigned)(uintptr_t)(void*)&Raw[0][0][0];
    const unsigned rawBase1 = (unsigned)(uintptr_t)(void*)&Raw[1][0][0];

    v8f acc[4];
    {
        v8f z = {0.f,0.f,0.f,0.f,0.f,0.f,0.f,0.f};
        acc[0] = z; acc[1] = z; acc[2] = z; acc[3] = z;
    }

    const int hi  = lane >> 4;
    const int row = m0 + (lane & 15);
    const float* Arow = A + (size_t)row * K;

    // conversion coordinates: thread t converts 8 consecutive floats of one row
    const int kS  = tid >> 3;                  // 0..31
    const int nbS = (tid & 7) * 8;             // 0,8,..,56
    const int hiK = (kS >> 4) << 4;

    // ---- prologue: tile0 -> Raw[0] (async), convert to Fsh[0]; tile1 -> Raw[1]
    stage_tile_async(W, N, n0, k0, rawBase0, tid);
    wait_async0();
    __syncthreads();
    convert_tile(Raw[0], Fsh[0], kS, nbS, hiK);
    if (k0 + TILE_K < k1)
        stage_tile_async(W, N, n0, k0 + TILE_K, rawBase1, tid);

    int p = 0, q = 1;
    for (int kk = k0; kk < k1; kk += TILE_K) {
        const bool more  = (kk +     TILE_K) < k1;
        const bool more2 = (kk + 2 * TILE_K) < k1;

        // Raw[q] landed; publish Fsh[p]; drain prior LDS reads (WAR safety)
        wait_async0();
        __syncthreads();

        // DMA tile kk+2 into the raw buffer we just finished converting
        if (more2)
            stage_tile_async(W, N, n0, kk + 2 * TILE_K, q ? rawBase0 : rawBase1, tid);

        // prefetch A two K-steps ahead
        if (kk + 2 * TILE_K < k1)
            __builtin_prefetch(Arow + kk + 2 * TILE_K, 0, 0);

        // ---- A fragment (16x32 bf16): lane<16 -> K {0..15},
        //      lane>=16 -> K {16..31}
        const float4* a0p = (const float4*)(Arow + kk + hi * 16);
        float4 a0 = a0p[0], a1 = a0p[1];
        const float4* a1p = (const float4*)(Arow + kk + 8 + hi * 16);
        float4 a2 = a1p[0], a3 = a1p[1];
        union { v16bf v; unsigned short u[16]; } af;
        af.u[0]  = f2bf(a0.x); af.u[1]  = f2bf(a0.y); af.u[2]  = f2bf(a0.z); af.u[3]  = f2bf(a0.w);
        af.u[4]  = f2bf(a1.x); af.u[5]  = f2bf(a1.y); af.u[6]  = f2bf(a1.z); af.u[7]  = f2bf(a1.w);
        af.u[8]  = f2bf(a2.x); af.u[9]  = f2bf(a2.y); af.u[10] = f2bf(a2.z); af.u[11] = f2bf(a2.w);
        af.u[12] = f2bf(a3.x); af.u[13] = f2bf(a3.y); af.u[14] = f2bf(a3.z); af.u[15] = f2bf(a3.w);

        // ---- B fragments + 4 WMMAs
        v16bf bfr[4];
        #pragma unroll
        for (int f = 0; f < 4; ++f)
            bfr[f] = *(const v16bf*)&Fsh[p][f][lane][0];
        #pragma unroll
        for (int f = 0; f < 4; ++f)
            acc[f] = __builtin_amdgcn_wmma_f32_16x16x32_bf16(
                         false, af.v, false, bfr[f], (short)0, acc[f], false, false);

        // ---- convert tile kk+1 (Raw[q]) -> Fsh[p^1] while DMA runs
        if (more)
            convert_tile(Raw[q], Fsh[p ^ 1], kS, nbS, hiK);

        p ^= 1; q ^= 1;
    }

    // ---- epilogue
    if (act < 0) {
        float* P = out0 + (size_t)blockIdx.z * ((size_t)M * N);
        #pragma unroll
        for (int f = 0; f < 4; ++f) {
            int col = n0 + f * 16 + (lane & 15);
            #pragma unroll
            for (int r = 0; r < 8; ++r) {
                int orow = m0 + r + hi * 8;
                P[(size_t)orow * N + col] = acc[f][r];
            }
        }
    } else {
        #pragma unroll
        for (int f = 0; f < 4; ++f) {
            int col = n0 + f * 16 + (lane & 15);
            float b = bias[col];
            #pragma unroll
            for (int r = 0; r < 8; ++r) {
                int orow = m0 + r + hi * 8;
                float v = acc[f][r] + b;
                v = act ? (1.0f / (1.0f + __expf(-v)))
                        : (v > 0.0f ? v : __expf(v) - 1.0f);
                float* o = (orow < split) ? (out0 + (size_t)orow * N + col)
                                          : (out1 + (size_t)(orow - split) * N + col);
                *o = v;
            }
        }
    }
}

// ---------------------------------------------------------------------------
// Split-K reduce for encoder FC1: h = elu(sum_c P[c] + be1)
// ---------------------------------------------------------------------------
__global__ __launch_bounds__(256) void reduce_bias_elu(
    const float* __restrict__ P, const float* __restrict__ bias,
    float* __restrict__ h)
{
    const size_t MH = (size_t)MROWS * HID;
    size_t i = (size_t)blockIdx.x * 256 + threadIdx.x;
    float s = P[i] + P[i + MH] + P[i + 2 * MH] + P[i + 3 * MH];
    float v = s + bias[i & (HID - 1)];
    h[i] = v > 0.f ? v : __expf(v) - 1.f;
}

// ---------------------------------------------------------------------------
// Encoder FC2: z = h(1024x512) @ We2(512x9) + be2. One wave per output col.
// ---------------------------------------------------------------------------
__global__ __launch_bounds__(288) void enc_fc2(
    const float* __restrict__ h, const float* __restrict__ We2,
    const float* __restrict__ be2,
    float* __restrict__ z_ws, float* __restrict__ z_out)
{
    int row  = blockIdx.x;
    int w    = threadIdx.x >> 5;
    int lane = threadIdx.x & 31;
    const float* hr = h + (size_t)row * HID;
    float s = 0.f;
    for (int k = lane; k < HID; k += 32) s += hr[k] * We2[k * 9 + w];
    #pragma unroll
    for (int off = 16; off > 0; off >>= 1) s += __shfl_xor(s, off, 32);
    if (lane == 0) {
        float v = s + be2[w];
        z_ws[row * 9 + w]  = v;
        z_out[row * 9 + w] = v;
    }
}

// ---------------------------------------------------------------------------
// SO(3) dynamics: moi inverse, pi_enc, 15-step Hamiltonian scan.
// ---------------------------------------------------------------------------
__global__ void dynamics(
    const float* __restrict__ z_ws,
    const float* __restrict__ moi_diag, const float* __restrict__ moi_off,
    float* __restrict__ z_pred_ws,
    float* __restrict__ z_pred_out, float* __restrict__ pi_enc_out,
    float* __restrict__ pi_pred_out)
{
    int b = threadIdx.x;               // blockDim.x == 64
    float d0 = moi_diag[0], d1 = moi_diag[1], d2 = moi_diag[2];
    float o0 = moi_off[0],  o1 = moi_off[1],  o2 = moi_off[2];

    float Mi[3][3];
    Mi[0][0] = d0*d0 + 1e-4f; Mi[0][1] = d0*o0;              Mi[0][2] = d0*o1;
    Mi[1][0] = Mi[0][1];      Mi[1][1] = o0*o0+d1*d1+1e-4f;  Mi[1][2] = o0*o1+d1*o2;
    Mi[2][0] = Mi[0][2];      Mi[2][1] = Mi[1][2];           Mi[2][2] = o1*o1+o2*o2+d2*d2+1e-4f;

    float c00 = Mi[1][1]*Mi[2][2] - Mi[1][2]*Mi[2][1];
    float c01 = Mi[1][2]*Mi[2][0] - Mi[1][0]*Mi[2][2];
    float c02 = Mi[1][0]*Mi[2][1] - Mi[1][1]*Mi[2][0];
    float det = Mi[0][0]*c00 + Mi[0][1]*c01 + Mi[0][2]*c02;
    float id  = 1.f / det;
    float Mo[3][3];
    Mo[0][0] = c00*id;
    Mo[0][1] = (Mi[0][2]*Mi[2][1] - Mi[0][1]*Mi[2][2])*id;
    Mo[0][2] = (Mi[0][1]*Mi[1][2] - Mi[0][2]*Mi[1][1])*id;
    Mo[1][0] = c01*id;
    Mo[1][1] = (Mi[0][0]*Mi[2][2] - Mi[0][2]*Mi[2][0])*id;
    Mo[1][2] = (Mi[0][2]*Mi[1][0] - Mi[0][0]*Mi[1][2])*id;
    Mo[2][0] = c02*id;
    Mo[2][1] = (Mi[0][1]*Mi[2][0] - Mi[0][0]*Mi[2][1])*id;
    Mo[2][2] = (Mi[0][0]*Mi[1][1] - Mi[0][1]*Mi[1][0])*id;

    const float* zb = z_ws + (size_t)b * SLEN * 9;
    float pi[3] = {0.f, 0.f, 0.f};

    const float inv2dt = 1.0f / (2.0f * DT_C);
    for (int t = 0; t < SLEN - 1; ++t) {
        const float* R1 = zb + t * 9;
        const float* R2 = zb + (t + 1) * 9;
        float A21=0, A12=0, A02=0, A20=0, A10=0, A01=0;
        #pragma unroll
        for (int j = 0; j < 3; ++j) {
            A21 += R1[j*3+2]*R2[j*3+1];  A12 += R1[j*3+1]*R2[j*3+2];
            A02 += R1[j*3+0]*R2[j*3+2];  A20 += R1[j*3+2]*R2[j*3+0];
            A10 += R1[j*3+1]*R2[j*3+0];  A01 += R1[j*3+0]*R2[j*3+1];
        }
        float w0 = (A21 - A12) * inv2dt;
        float w1 = (A02 - A20) * inv2dt;
        float w2 = (A10 - A01) * inv2dt;
        float p0 = Mo[0][0]*w0 + Mo[0][1]*w1 + Mo[0][2]*w2;
        float p1 = Mo[1][0]*w0 + Mo[1][1]*w1 + Mo[1][2]*w2;
        float p2 = Mo[2][0]*w0 + Mo[2][1]*w1 + Mo[2][2]*w2;
        float* pe = pi_enc_out + ((size_t)b * 15 + t) * 3;
        pe[0] = p0; pe[1] = p1; pe[2] = p2;
        if (t == 0) { pi[0] = p0; pi[1] = p1; pi[2] = p2; }
    }

    float R[9];
    #pragma unroll
    for (int i = 0; i < 9; ++i) R[i] = zb[i];

    {
        float* zo = z_pred_out + (size_t)(b * SLEN) * 9;
        float* zw = z_pred_ws  + (size_t)(b * SLEN) * 9;
        #pragma unroll
        for (int i = 0; i < 9; ++i) { zo[i] = R[i]; zw[i] = R[i]; }
        float* pp = pi_pred_out + (size_t)(b * SLEN) * 3;
        pp[0] = pi[0]; pp[1] = pi[1]; pp[2] = pi[2];
    }

    for (int t = 1; t < SLEN; ++t) {
        float g0 = Mi[0][0]*pi[0] + Mi[0][1]*pi[1] + Mi[0][2]*pi[2];
        float g1 = Mi[1][0]*pi[0] + Mi[1][1]*pi[1] + Mi[1][2]*pi[2];
        float g2 = Mi[2][0]*pi[0] + Mi[2][1]*pi[1] + Mi[2][2]*pi[2];
        float x = g0 * DT_C, y = g1 * DT_C, z = g2 * DT_C;
        float th = sqrtf(x*x + y*y + z*z);
        float a, bb;
        if (th < 1e-8f) { a = 1.f; bb = 0.5f; }
        else            { a = __sinf(th)/th; bb = (1.f - __cosf(th))/(th*th); }
        float E[9];
        E[0] = 1.f + bb*(-(y*y + z*z)); E[1] = -a*z + bb*(x*y);        E[2] =  a*y + bb*(x*z);
        E[3] =  a*z + bb*(x*y);         E[4] = 1.f + bb*(-(x*x+z*z));  E[5] = -a*x + bb*(y*z);
        E[6] = -a*y + bb*(x*z);         E[7] =  a*x + bb*(y*z);        E[8] = 1.f + bb*(-(x*x+y*y));
        float Rn[9];
        #pragma unroll
        for (int i = 0; i < 3; ++i)
            #pragma unroll
            for (int j = 0; j < 3; ++j)
                Rn[i*3+j] = R[i*3+0]*E[0*3+j] + R[i*3+1]*E[1*3+j] + R[i*3+2]*E[2*3+j];
        float pn0 = pi[0] + DT_C * (pi[1]*g2 - pi[2]*g1);
        float pn1 = pi[1] + DT_C * (pi[2]*g0 - pi[0]*g2);
        float pn2 = pi[2] + DT_C * (pi[0]*g1 - pi[1]*g0);
        #pragma unroll
        for (int i = 0; i < 9; ++i) R[i] = Rn[i];
        pi[0] = pn0; pi[1] = pn1; pi[2] = pn2;

        float* zo = z_pred_out + (size_t)(b * SLEN + t) * 9;
        float* zw = z_pred_ws  + (size_t)(b * SLEN + t) * 9;
        #pragma unroll
        for (int i = 0; i < 9; ++i) { zo[i] = R[i]; zw[i] = R[i]; }
        float* pp = pi_pred_out + (size_t)(b * SLEN + t) * 3;
        pp[0] = pi[0]; pp[1] = pi[1]; pp[2] = pi[2];
    }
}

// ---------------------------------------------------------------------------
// Decoder FC1: hd = elu(z(2048x9) @ Wd1(9x512) + bd1). K=9: plain VALU.
// ---------------------------------------------------------------------------
__global__ __launch_bounds__(256) void dec_fc1(
    const float* __restrict__ z, const float* __restrict__ Wd1,
    const float* __restrict__ bd1, float* __restrict__ hd)
{
    int row = blockIdx.x;
    const float* zr = z + (size_t)row * 9;
    float zv[9];
    #pragma unroll
    for (int k = 0; k < 9; ++k) zv[k] = zr[k];
    for (int n = threadIdx.x; n < HID; n += 256) {
        float s = bd1[n];
        #pragma unroll
        for (int k = 0; k < 9; ++k) s += zv[k] * Wd1[k * HID + n];
        hd[(size_t)row * HID + n] = s > 0.f ? s : __expf(s) - 1.f;
    }
}

// ---------------------------------------------------------------------------
extern "C" void kernel_launch(void* const* d_in, const int* in_sizes, int n_in,
                              void* d_out, int out_size, void* d_ws, size_t ws_size,
                              hipStream_t stream) {
    const float* x        = (const float*)d_in[0];
    const float* moi_diag = (const float*)d_in[1];
    const float* moi_off  = (const float*)d_in[2];
    const float* We1      = (const float*)d_in[3];
    const float* be1      = (const float*)d_in[4];
    const float* We2      = (const float*)d_in[5];
    const float* be2      = (const float*)d_in[6];
    const float* Wd1      = (const float*)d_in[7];
    const float* bd1      = (const float*)d_in[8];
    const float* Wd2      = (const float*)d_in[9];
    const float* bd2      = (const float*)d_in[10];
    (void)in_sizes; (void)n_in; (void)out_size; (void)ws_size;

    float* out = (float*)d_out;
    float* ws  = (float*)d_ws;
    float* ws_h    = ws + WS_H;
    float* ws_z    = ws + WS_Z;
    float* ws_hd   = ws + WS_HD;
    float* ws_part = ws + WS_PART;

    // 1) Encoder FC1 split-K: partials P[c] = x @ We1 over K-chunk c
    {
        dim3 grid(MROWS / TILE_M, HID / TILE_N, KSPLIT);   // 8 x 8 x 4
        wmma_gemm<<<grid, 256, 0, stream>>>(
            x, We1, be1, ws_part, ws_part,
            MROWS, HID, DIM, KCHUNK, MROWS, /*act=*/-1);
    }
    // 1b) h = elu(sum_c P[c] + be1)
    reduce_bias_elu<<<(MROWS * HID) / 256, 256, 0, stream>>>(ws_part, be1, ws_h);

    // 2) Encoder FC2: z_enc = h @ We2 + be2
    enc_fc2<<<MROWS, 288, 0, stream>>>(ws_h, We2, be2, ws_z, out + OUT_ZENC);

    // 3) SO(3) dynamics -> z_pred, pi_enc, pi_pred
    dynamics<<<1, BSZ, 0, stream>>>(
        ws_z, moi_diag, moi_off, ws_z + (size_t)MROWS * 9,
        out + OUT_ZPRED, out + OUT_PIENC, out + OUT_PIPRED);

    // 4) Decoder FC1: hd = elu(z_all @ Wd1 + bd1)
    dec_fc1<<<DROWS, 256, 0, stream>>>(ws_z, Wd1, bd1, ws_hd);

    // 5) Decoder FC2: xhat = sigmoid(hd @ Wd2 + bd2), fused recon/pred split
    {
        dim3 grid(DROWS / TILE_M, DIM / TILE_N, 1);        // 16 x 588
        wmma_gemm<<<grid, 256, 0, stream>>>(
            ws_hd, Wd2, bd2, out + OUT_RECON, out + OUT_PRED,
            DROWS, DIM, HID, HID, MROWS, /*act=*/1);
    }
}
